// PagedAttention_54554674594340
// MI455X (gfx1250) — matrix operand order repaired and checked by reading, not compiled, not run on hardware
//
#include <hip/hip_runtime.h>
#include <hip/hip_bf16.h>
#include <math.h>

typedef _Float16 v8h  __attribute__((ext_vector_type(8)));
typedef _Float16 v16h __attribute__((ext_vector_type(16)));
typedef float    v8f  __attribute__((ext_vector_type(8)));

#define E_DIM  4096
#define B_DIM  8
#define H_DIM  32
#define D_DIM  128
#define T_TOT  2048
#define T_PREV 2047
#define KSPLIT 4
#define KCHUNK (E_DIM / KSPLIT)          // 1024
#define YSZ    (B_DIM * E_DIM)           // 32768 outputs per GEMM

// ---------------------------------------------------------------- f32 -> f16
__global__ void cvt_f32_to_f16(const float* __restrict__ src,
                               _Float16* __restrict__ dst, int n) {
    int i = blockIdx.x * blockDim.x + threadIdx.x;
    if (i < n) dst[i] = (_Float16)src[i];
}

// ------------------------------------------- WMMA GEMM partials: X(8xE)@W^T
// grid = (64, nMat, KSPLIT); block = 128 threads = 4 waves, 16 cols per wave.
// Each block accumulates one K-chunk of 1024 and writes an (8,4096) partial.
__global__ __launch_bounds__(128)
void gemm8_wmma_part(const _Float16* __restrict__ xh,   // (8, 4096) f16
                     const float* __restrict__ W0,      // (4096,4096) f32 each
                     const float* __restrict__ W1,
                     const float* __restrict__ W2,
                     float* __restrict__ part) {        // (nMat,KSPLIT,8,4096)
    const int lane = threadIdx.x & 31;
    const int wave = threadIdx.x >> 5;
    const int n    = blockIdx.x * 64 + wave * 16 + (lane & 15);
    const int hi   = (lane >> 4) & 1;
    const int mat  = blockIdx.y;
    const int kz   = blockIdx.z;
    const float* W = (mat == 0) ? W0 : ((mat == 1) ? W1 : W2);

    // A fragment: lane holds row m = lane&15 (rows 8..15 masked to zero)
    const int m = lane & 15;
    const _Float16* xrow = xh + (size_t)(m < 8 ? m : 0) * E_DIM;
    const _Float16 amask = (m < 8) ? (_Float16)1.0f : (_Float16)0.0f;
    const int kbA = hi * 8;      // lanes 0-15: K 0-7,16-23 ; lanes 16-31: K 8-15,24-31

    // B fragment: lane holds col n; lanes 0-15: K 0-7,16-23 ; lanes 16-31: K 8-15,24-31
    const float* wrow = W + (size_t)n * E_DIM;
    const int kbB = hi * 8;

    const int k0 = kz * KCHUNK;
    v8f acc = {};
    for (int kk = k0; kk < k0 + KCHUNK; kk += 32) {
        // ---- A fragment (two b128 loads of 8 halves)
        v8h alo = *(const v8h*)(xrow + kk + kbA);
        v8h ahi = *(const v8h*)(xrow + kk + 16 + kbA);
        v16h a;
        #pragma unroll
        for (int i = 0; i < 8; ++i) { a[i] = alo[i]; a[i + 8] = ahi[i]; }
        a *= amask;

        // ---- B fragment (four b128 f32 loads, convert in-register)
        const float4* wp = (const float4*)(wrow + kk + kbB);
        float wf[16];
        *(float4*)(wf + 0)  = wp[0];
        *(float4*)(wf + 4)  = wp[1];
        *(float4*)(wf + 8)  = wp[4];
        *(float4*)(wf + 12) = wp[5];
        v16h b;
        #pragma unroll
        for (int i = 0; i < 16; ++i) b[i] = (_Float16)wf[i];

        acc = __builtin_amdgcn_wmma_f32_16x16x32_f16(
                  false, a, false, b, (short)0, acc, false, false);

        __builtin_prefetch(wrow + kk + 512, 0, 1);   // global_prefetch_b8
    }

    // D layout: lanes 0-15 hold M=0..7 in acc[0..7]
    if (lane < 16) {
        float* dst = part + (size_t)(mat * KSPLIT + kz) * YSZ;
        #pragma unroll
        for (int j = 0; j < 8; ++j)
            dst[(size_t)j * E_DIM + n] = acc[j];
    }
}

// ------------------------------------------ partial reduce (+bias), dense out
__global__ __launch_bounds__(256)
void reduce_dense(const float* __restrict__ part, const float* __restrict__ bias,
                  float* __restrict__ Y) {
    int r = blockIdx.x * blockDim.x + threadIdx.x;
    if (r >= YSZ) return;
    float s = bias[r & (E_DIM - 1)];
    #pragma unroll
    for (int z = 0; z < KSPLIT; ++z) s += part[(size_t)z * YSZ + r];
    Y[r] = s;
}

// --------------------------- partial reduce (+bias), scatter into cache t=2047
__global__ __launch_bounds__(256)
void reduce_scatter(const float* __restrict__ part, const float* __restrict__ bias,
                    float* __restrict__ Yc) {
    int r = blockIdx.x * blockDim.x + threadIdx.x;
    if (r >= YSZ) return;
    int n = r & (E_DIM - 1), m = r >> 12;
    float s = bias[n];
    #pragma unroll
    for (int z = 0; z < KSPLIT; ++z) s += part[(size_t)z * YSZ + r];
    int h = n >> 7, dd = n & 127;
    Yc[((size_t)(m * H_DIM + h) * T_TOT + (T_TOT - 1)) * D_DIM + dd] = s;
}

// ---------------------------------------------------- cache body copy (b128)
// grid = (256, 256, 2): x covers the row body, y = (b,h) row, z = k/v select.
__global__ __launch_bounds__(256)
void cache_copy(const float4* __restrict__ ksrc, const float4* __restrict__ vsrc,
                float4* __restrict__ kdst, float4* __restrict__ vdst) {
    const int NROW4 = T_PREV * D_DIM / 4;   // 65504
    const int DROW4 = T_TOT  * D_DIM / 4;   // 65536
    int w = blockIdx.x * blockDim.x + threadIdx.x;
    if (w >= NROW4) return;
    int r = blockIdx.y;
    size_t s = (size_t)r * NROW4 + w;
    size_t d = (size_t)r * DROW4 + w;
    if (blockIdx.z == 0) kdst[d] = ksrc[s];
    else                 vdst[d] = vsrc[s];
}

// -------------------------------------------------- attention for one (b,h)
// Wave-cooperative score rows: one wave reads one K row as a coalesced 512B
// burst, butterfly-reduces the dot product, softmax in LDS, coalesced V pass.
__global__ __launch_bounds__(256)
void attn_decode(const float* __restrict__ qbuf,     // (8, 4096) f32
                 const float* __restrict__ kc,       // (8,32,2048,128)
                 const float* __restrict__ vc,       // (8,32,2048,128)
                 _Float16* __restrict__ attn_h) {    // (8, 4096) f16
    __shared__ float sc[T_TOT];
    __shared__ float red[256];

    const int tid = threadIdx.x;
    const int wv  = tid >> 5, ln = tid & 31;
    const int bh  = blockIdx.x;                      // b*32 + h
    const float* kcb = kc + (size_t)bh * T_TOT * D_DIM;
    const float* vcb = vc + (size_t)bh * T_TOT * D_DIM;

    // q slice for this lane (16B), shared by all 8 waves via L0/L2
    float4 q4 = ((const float4*)(qbuf + (size_t)bh * D_DIM))[ln];

    const float scale = 0.08838834764831845f;        // 1/sqrt(128)
    float lmax = -3.4e38f;
    for (int t = wv; t < T_TOT; t += 8) {
        float4 k4 = ((const float4*)(kcb + (size_t)t * D_DIM))[ln];
        float p = k4.x * q4.x + k4.y * q4.y + k4.z * q4.z + k4.w * q4.w;
        #pragma unroll
        for (int msk = 16; msk; msk >>= 1) p += __shfl_xor(p, msk, 32);
        p *= scale;
        if (ln == 0) sc[t] = p;
        lmax = fmaxf(lmax, p);
    }
    red[tid] = lmax; __syncthreads();
    for (int s = 128; s > 0; s >>= 1) {
        if (tid < s) red[tid] = fmaxf(red[tid], red[tid + s]);
        __syncthreads();
    }
    float mx = red[0];
    __syncthreads();

    float lsum = 0.f;
    for (int t = tid; t < T_TOT; t += 256) {
        float p = __expf(sc[t] - mx);
        sc[t] = p;
        lsum += p;
    }
    red[tid] = lsum; __syncthreads();
    for (int s = 128; s > 0; s >>= 1) {
        if (tid < s) red[tid] += red[tid + s];
        __syncthreads();
    }
    float inv = 1.0f / red[0];
    __syncthreads();

    const int d = tid & 127, half = tid >> 7;
    float acc = 0.f;
    for (int t = half; t < T_TOT; t += 2)            // coalesced across lanes
        acc += sc[t] * vcb[(size_t)t * D_DIM + d];
    red[tid] = acc; __syncthreads();
    if (tid < D_DIM) {
        float o = (red[tid] + red[tid + 128]) * inv;
        attn_h[(size_t)bh * D_DIM + tid] = (_Float16)o;
    }
}

// ---------------------------------------------------------------------------
extern "C" void kernel_launch(void* const* d_in, const int* in_sizes, int n_in,
                              void* d_out, int out_size, void* d_ws, size_t ws_size,
                              hipStream_t stream) {
    (void)in_sizes; (void)n_in; (void)out_size; (void)ws_size;

    const float* x       = (const float*)d_in[0];
    const float* k_cache = (const float*)d_in[1];
    const float* v_cache = (const float*)d_in[2];
    const float* Wq = (const float*)d_in[3];
    const float* bq = (const float*)d_in[4];
    const float* Wk = (const float*)d_in[5];
    const float* bk = (const float*)d_in[6];
    const float* Wv = (const float*)d_in[7];
    const float* bv = (const float*)d_in[8];
    const float* Wo = (const float*)d_in[9];
    const float* bo = (const float*)d_in[10];

    // output tuple layout: out (8*4096) | key_cache (8*32*2048*128) | value_cache
    float* out = (float*)d_out;
    float* kc  = out + (size_t)YSZ;
    float* vc  = kc + (size_t)B_DIM * H_DIM * T_TOT * D_DIM;

    // workspace: xh f16 64KB | q f32 128KB | attn_h f16 64KB | partials 1.5MB
    _Float16* xh     = (_Float16*)d_ws;
    float*    qbuf   = (float*)((char*)d_ws + 65536);
    _Float16* attn_h = (_Float16*)((char*)d_ws + 65536 + 131072);
    float*    part   = (float*)((char*)d_ws + 65536 + 131072 + 65536);

    cvt_f32_to_f16<<<(YSZ + 255) / 256, 256, 0, stream>>>(x, xh, YSZ);

    // fused QKV partial GEMMs: 64 N-tiles x 3 matrices x 4 K-chunks = 768 blocks
    gemm8_wmma_part<<<dim3(E_DIM / 64, 3, KSPLIT), 128, 0, stream>>>(
        xh, Wq, Wk, Wv, part);
    reduce_dense  <<<(YSZ + 255) / 256, 256, 0, stream>>>(part,              bq, qbuf);
    reduce_scatter<<<(YSZ + 255) / 256, 256, 0, stream>>>(part + 1L*KSPLIT*YSZ, bk, kc);
    reduce_scatter<<<(YSZ + 255) / 256, 256, 0, stream>>>(part + 2L*KSPLIT*YSZ, bv, vc);

    cache_copy<<<dim3(256, B_DIM * H_DIM, 2), 256, 0, stream>>>(
        (const float4*)k_cache, (const float4*)v_cache, (float4*)kc, (float4*)vc);

    attn_decode<<<B_DIM * H_DIM, 256, 0, stream>>>(qbuf, kc, vc, attn_h);

    // output projection: same kernel, 1 matrix x 4 K-chunks
    gemm8_wmma_part<<<dim3(E_DIM / 64, 1, KSPLIT), 128, 0, stream>>>(
        attn_h, Wo, Wo, Wo, part);
    reduce_dense<<<(YSZ + 255) / 256, 256, 0, stream>>>(part, bo, out);
}
